// Attention_27857157882270
// MI455X (gfx1250) — compile-verified, matrix operand order checked by reading, not run on hardware
//
#include <hip/hip_runtime.h>
#include <stdint.h>

// ---------------------------------------------------------------------------
// Attention: Q=xWq+b, K=xWk+b, V=xWv+b, softmax(QK^T/sqrt(D) masked) @ V
// B=8, S=2048, D=512.
// GEMMs on v_wmma_f32_16x16x32_bf16. Big GEMMs stage operand tiles in LDS
// with global_load_async_to_lds_b128 (ASYNCcnt) + double buffering.
// ---------------------------------------------------------------------------

#define S_LEN 2048
#define D_DIM 512
#define B_N   8
#define INV_SQRT_D 0.04419417382415922f   // 1/sqrt(512)

#define BM 64        // block tile M
#define BN 128       // block tile N
#define BK 32        // K step
#define LDS_PITCH 40 // bf16 elements per LDS tile row (80B) -> conflict-free

typedef __attribute__((ext_vector_type(16))) __bf16 v16bf;
typedef __attribute__((ext_vector_type(8)))  float  v8f;

union BF16Frag { v16bf v; uint32_t u[8]; };
union AccFrag  { v8f  v; float    f[8]; };

__device__ __forceinline__ uint16_t bf16bits(float x) {
    __bf16 h = (__bf16)x;                       // hardware cvt
    return __builtin_bit_cast(uint16_t, h);
}

__device__ __forceinline__ uint32_t pack_bf16(float lo, float hi) {
    return (uint32_t)bf16bits(lo) | ((uint32_t)bf16bits(hi) << 16);
}

__device__ __forceinline__ v8f zero8() {
    v8f c;
    #pragma unroll
    for (int i = 0; i < 8; ++i) c[i] = 0.0f;
    return c;
}

// 16x32 bf16 fragment from a row-major bf16 matrix (global memory path).
// CDNA5 16-bit A layout: lane<16 -> row=lane, K={0..7,16..23};
// lane>=16 -> row=lane-16, K={8..15,24..31}: two b128 loads per lane.
__device__ __forceinline__ v16bf load_frag_bf16(const uint16_t* p, int ld) {
    const int lane = threadIdx.x & 31;
    const int r = lane & 15;
    const int c = (lane >> 4) << 3;
    const uint32_t* q0 = reinterpret_cast<const uint32_t*>(p + (size_t)r * ld + c);
    const uint32_t* q1 = reinterpret_cast<const uint32_t*>(p + (size_t)r * ld + c + 16);
    BF16Frag f;
    #pragma unroll
    for (int i = 0; i < 4; ++i) { f.u[i] = q0[i]; f.u[4 + i] = q1[i]; }
    return f.v;
}

// Same fragment out of an LDS tile with LDS_PITCH row pitch (ds_load_b128 x2).
__device__ __forceinline__ v16bf load_frag_lds(const uint16_t* p) {
    const int lane = threadIdx.x & 31;
    const int r = lane & 15;
    const int c = (lane >> 4) << 3;
    const uint32_t* q0 = reinterpret_cast<const uint32_t*>(p + r * LDS_PITCH + c);
    const uint32_t* q1 = reinterpret_cast<const uint32_t*>(p + r * LDS_PITCH + c + 16);
    BF16Frag f;
    #pragma unroll
    for (int i = 0; i < 4; ++i) { f.u[i] = q0[i]; f.u[4 + i] = q1[i]; }
    return f.v;
}

// Cooperative async copy of a (rows x 32) bf16 tile (row-major, leading dim
// ldg) into an LDS tile with LDS_PITCH pitch. 16B per lane per issue;
// tracked by ASYNCcnt, completion via s_wait_asynccnt.
__device__ __forceinline__ void async_copy_tile(const uint16_t* g, int ldg,
                                                uint16_t* l, int rows) {
    const int tid = threadIdx.x;
    const int chunks = rows * 4;                 // 4 x 16B per 32-elem row
    #pragma unroll
    for (int c = tid; c < chunks; c += 128) {
        const int row = c >> 2;
        const int col = (c & 3) * 8;             // elements
        const uint64_t ga = (uint64_t)(uintptr_t)(g + (size_t)row * ldg + col);
        const uint32_t la = (uint32_t)(uintptr_t)(l + row * LDS_PITCH + col);
        asm volatile("global_load_async_to_lds_b128 %0, %1, off"
                     :: "v"(la), "v"(ga) : "memory");
    }
}

__device__ __forceinline__ void wait_async_all() {
    asm volatile("s_wait_asynccnt 0x0" ::: "memory");
}

#define WMMA_BF16(A, B, C) \
    __builtin_amdgcn_wmma_f32_16x16x32_bf16(false, (A), false, (B), (short)0, (C), false, false)

// ---------------------------------------------------------------------------
// Kernel 0: f32 -> bf16 streaming convert (8 elements/thread, b128 traffic)
// ---------------------------------------------------------------------------
__global__ __launch_bounds__(256) void cvt_bf16_kernel(
        const float* __restrict__ src, uint16_t* __restrict__ dst, int n8) {
    const int i = blockIdx.x * 256 + threadIdx.x;
    if (i >= n8) return;
    const size_t base = (size_t)i * 8;
    uint32_t out[4];
    #pragma unroll
    for (int j = 0; j < 4; ++j)
        out[j] = pack_bf16(src[base + 2 * j], src[base + 2 * j + 1]);
    uint32_t* d = reinterpret_cast<uint32_t*>(dst + base);
    #pragma unroll
    for (int j = 0; j < 4; ++j) d[j] = out[j];
}

// ---------------------------------------------------------------------------
// Kernel 1: projection  out = x @ W^T + b  (M=16384, N=K=512), bf16 operands,
// direct-global fragments (compiler software-pipelines the b128 loads).
// 4 waves/block (2x2), wave tile 32x64 -> block tile 64x128.
// ---------------------------------------------------------------------------
__global__ __launch_bounds__(128) void qkv_proj_kernel(
        const uint16_t* __restrict__ xbf, const uint16_t* __restrict__ Wbf,
        const float* __restrict__ bias, uint16_t* __restrict__ out,
        int transposeOut) {
    const int wave = threadIdx.x >> 5;
    const int m0 = blockIdx.y * BM + (wave >> 1) * 32;
    const int n0 = blockIdx.x * BN + (wave & 1) * 64;

    v8f cc[2][4];
    #pragma unroll
    for (int i = 0; i < 2; ++i)
        #pragma unroll
        for (int j = 0; j < 4; ++j) cc[i][j] = zero8();

    for (int k = 0; k < D_DIM; k += BK) {
        if (k + BK < D_DIM) {   // global_prefetch_b8 of next K-slab
            __builtin_prefetch(xbf + (size_t)m0 * D_DIM + k + BK, 0, 1);
            __builtin_prefetch(Wbf + (size_t)n0 * D_DIM + k + BK, 0, 1);
        }
        v16bf a0 = load_frag_bf16(xbf + (size_t)m0 * D_DIM + k, D_DIM);
        v16bf a1 = load_frag_bf16(xbf + (size_t)(m0 + 16) * D_DIM + k, D_DIM);
        v16bf bfr[4];
        #pragma unroll
        for (int j = 0; j < 4; ++j)
            bfr[j] = load_frag_bf16(Wbf + (size_t)(n0 + 16 * j) * D_DIM + k, D_DIM);
        #pragma unroll
        for (int j = 0; j < 4; ++j) {
            cc[0][j] = WMMA_BF16(a0, bfr[j], cc[0][j]);
            cc[1][j] = WMMA_BF16(a1, bfr[j], cc[1][j]);
        }
    }

    const int lane = threadIdx.x & 31;
    const int ln = lane & 15;
    const int lm = (lane >> 4) << 3;
    #pragma unroll
    for (int ti = 0; ti < 2; ++ti) {
        #pragma unroll
        for (int tj = 0; tj < 4; ++tj) {
            AccFrag af; af.v = cc[ti][tj];
            const int n = n0 + tj * 16 + ln;
            const float bv = bias[n];
            #pragma unroll
            for (int r = 0; r < 8; ++r) {
                const int m = m0 + ti * 16 + lm + r;
                const float val = af.f[r] + bv;
                size_t idx;
                if (!transposeOut) {
                    idx = (size_t)m * D_DIM + n;
                } else {
                    const int bb = m / S_LEN, s = m % S_LEN;
                    idx = ((size_t)bb * D_DIM + n) * S_LEN + s;
                }
                out[idx] = bf16bits(val);
            }
        }
    }
}

// ---------------------------------------------------------------------------
// Kernel 2: logits = (Q @ K^T)/sqrt(D), masked (per batch 2048x2048x512).
// Async-LDS double-buffered operand staging.
// ---------------------------------------------------------------------------
__global__ __launch_bounds__(128) void scores_kernel(
        const uint16_t* __restrict__ Qbf, const uint16_t* __restrict__ Kbf,
        const int* __restrict__ mask, float* __restrict__ logits) {
    __shared__ uint16_t As[2][BM * LDS_PITCH];   // 2 x 5KB
    __shared__ uint16_t Bs[2][BN * LDS_PITCH];   // 2 x 10KB
    const int b = blockIdx.z;
    const int mBlk = blockIdx.y * BM;
    const int nBlk = blockIdx.x * BN;
    const uint16_t* Qb = Qbf + (size_t)b * S_LEN * D_DIM + (size_t)mBlk * D_DIM;
    const uint16_t* Kb = Kbf + (size_t)b * S_LEN * D_DIM + (size_t)nBlk * D_DIM;

    const int wave = threadIdx.x >> 5;
    const int wm = (wave >> 1) * 32;   // A sub-rows within block tile
    const int wn = (wave & 1) * 64;    // B sub-rows (N) within block tile

    v8f cc[2][4];
    #pragma unroll
    for (int i = 0; i < 2; ++i)
        #pragma unroll
        for (int j = 0; j < 4; ++j) cc[i][j] = zero8();

    async_copy_tile(Qb, D_DIM, &As[0][0], BM);
    async_copy_tile(Kb, D_DIM, &Bs[0][0], BN);

    for (int k = 0, it = 0; k < D_DIM; k += BK, ++it) {
        const int cur = it & 1;
        wait_async_all();
        __syncthreads();
        if (k + BK < D_DIM) {
            async_copy_tile(Qb + k + BK, D_DIM, &As[cur ^ 1][0], BM);
            async_copy_tile(Kb + k + BK, D_DIM, &Bs[cur ^ 1][0], BN);
        }
        v16bf a0 = load_frag_lds(&As[cur][(wm) * LDS_PITCH]);
        v16bf a1 = load_frag_lds(&As[cur][(wm + 16) * LDS_PITCH]);
        v16bf bfr[4];
        #pragma unroll
        for (int j = 0; j < 4; ++j)
            bfr[j] = load_frag_lds(&Bs[cur][(wn + 16 * j) * LDS_PITCH]);
        #pragma unroll
        for (int j = 0; j < 4; ++j) {
            cc[0][j] = WMMA_BF16(a0, bfr[j], cc[0][j]);
            cc[1][j] = WMMA_BF16(a1, bfr[j], cc[1][j]);
        }
    }

    const int lane = threadIdx.x & 31;
    const int ln = lane & 15;
    const int lm = (lane >> 4) << 3;
    #pragma unroll
    for (int ti = 0; ti < 2; ++ti) {
        #pragma unroll
        for (int tj = 0; tj < 4; ++tj) {
            AccFrag af; af.v = cc[ti][tj];
            const int n = nBlk + wn + tj * 16 + ln;
            #pragma unroll
            for (int r = 0; r < 8; ++r) {
                const int m = mBlk + wm + ti * 16 + lm + r;
                const size_t idx = ((size_t)b * S_LEN + m) * S_LEN + n;
                float val = af.f[r] * INV_SQRT_D;
                if (mask[idx] == 0) val = -1e9f;
                logits[idx] = val;
            }
        }
    }
}

// ---------------------------------------------------------------------------
// Kernel 3: in-place row softmax over 2048; also emits bf16 weights.
// ---------------------------------------------------------------------------
__global__ __launch_bounds__(256) void softmax_kernel(
        float* __restrict__ w, uint16_t* __restrict__ wbf) {
    const size_t row = blockIdx.x;
    float* p = w + row * S_LEN;
    uint16_t* q = wbf + row * S_LEN;
    const int t = threadIdx.x;
    const int lane = t & 31;
    const int wv = t >> 5;
    __shared__ float smax[8], ssum[8];

    float v[8];
    float mx = -3.4e38f;
    #pragma unroll
    for (int i = 0; i < 8; ++i) { v[i] = p[t + 256 * i]; mx = fmaxf(mx, v[i]); }
    #pragma unroll
    for (int off = 16; off > 0; off >>= 1) mx = fmaxf(mx, __shfl_xor(mx, off, 32));
    if (lane == 0) smax[wv] = mx;
    __syncthreads();
    float rowmax = smax[0];
    #pragma unroll
    for (int i = 1; i < 8; ++i) rowmax = fmaxf(rowmax, smax[i]);

    float s = 0.0f;
    #pragma unroll
    for (int i = 0; i < 8; ++i) { v[i] = __expf(v[i] - rowmax); s += v[i]; }
    #pragma unroll
    for (int off = 16; off > 0; off >>= 1) s += __shfl_xor(s, off, 32);
    if (lane == 0) ssum[wv] = s;
    __syncthreads();
    float total = 0.0f;
    #pragma unroll
    for (int i = 0; i < 8; ++i) total += ssum[i];

    const float inv = 1.0f / total;
    #pragma unroll
    for (int i = 0; i < 8; ++i) {
        const float o = v[i] * inv;
        p[t + 256 * i] = o;
        q[t + 256 * i] = bf16bits(o);
    }
}

// ---------------------------------------------------------------------------
// Kernel 4: Sout = weights @ V (per batch 2048x512x2048), bf16 operands,
// async-LDS double-buffered staging (64 K-steps).
// ---------------------------------------------------------------------------
__global__ __launch_bounds__(128) void out_kernel(
        const uint16_t* __restrict__ Wbf, const uint16_t* __restrict__ Vt,
        float* __restrict__ Sout) {
    __shared__ uint16_t As[2][BM * LDS_PITCH];
    __shared__ uint16_t Bs[2][BN * LDS_PITCH];
    const int b = blockIdx.z;
    const int mBlk = blockIdx.y * BM;     // query rows
    const int nBlk = blockIdx.x * BN;     // d columns
    const uint16_t* Wb = Wbf + (size_t)b * S_LEN * S_LEN + (size_t)mBlk * S_LEN;
    const uint16_t* Vtb = Vt + (size_t)b * D_DIM * S_LEN + (size_t)nBlk * S_LEN;

    const int wave = threadIdx.x >> 5;
    const int wm = (wave >> 1) * 32;
    const int wn = (wave & 1) * 64;

    v8f cc[2][4];
    #pragma unroll
    for (int i = 0; i < 2; ++i)
        #pragma unroll
        for (int j = 0; j < 4; ++j) cc[i][j] = zero8();

    async_copy_tile(Wb, S_LEN, &As[0][0], BM);
    async_copy_tile(Vtb, S_LEN, &Bs[0][0], BN);

    for (int k = 0, it = 0; k < S_LEN; k += BK, ++it) {
        const int cur = it & 1;
        wait_async_all();
        __syncthreads();
        if (k + BK < S_LEN) {
            async_copy_tile(Wb + k + BK, S_LEN, &As[cur ^ 1][0], BM);
            async_copy_tile(Vtb + k + BK, S_LEN, &Bs[cur ^ 1][0], BN);
        }
        v16bf a0 = load_frag_lds(&As[cur][(wm) * LDS_PITCH]);
        v16bf a1 = load_frag_lds(&As[cur][(wm + 16) * LDS_PITCH]);
        v16bf bfr[4];
        #pragma unroll
        for (int j = 0; j < 4; ++j)
            bfr[j] = load_frag_lds(&Bs[cur][(wn + 16 * j) * LDS_PITCH]);
        #pragma unroll
        for (int j = 0; j < 4; ++j) {
            cc[0][j] = WMMA_BF16(a0, bfr[j], cc[0][j]);
            cc[1][j] = WMMA_BF16(a1, bfr[j], cc[1][j]);
        }
    }

    const int lane = threadIdx.x & 31;
    const int ln = lane & 15;
    const int lm = (lane >> 4) << 3;
    #pragma unroll
    for (int ti = 0; ti < 2; ++ti) {
        #pragma unroll
        for (int tj = 0; tj < 4; ++tj) {
            AccFrag af; af.v = cc[ti][tj];
            const int n = nBlk + wn + tj * 16 + ln;
            #pragma unroll
            for (int r = 0; r < 8; ++r) {
                const int m = mBlk + wm + ti * 16 + lm + r;
                Sout[((size_t)b * S_LEN + m) * D_DIM + n] = af.f[r];
            }
        }
    }
}

// ---------------------------------------------------------------------------
extern "C" void kernel_launch(void* const* d_in, const int* in_sizes, int n_in,
                              void* d_out, int out_size, void* d_ws, size_t ws_size,
                              hipStream_t stream) {
    (void)in_sizes; (void)n_in; (void)out_size; (void)ws_size;

    const float* x    = (const float*)d_in[0];
    const int*   mask = (const int*)d_in[1];
    const float* Wq   = (const float*)d_in[2];
    const float* bq   = (const float*)d_in[3];
    const float* Wk   = (const float*)d_in[4];
    const float* bk   = (const float*)d_in[5];
    const float* Wv   = (const float*)d_in[6];
    const float* bv   = (const float*)d_in[7];

    float* Sout    = (float*)d_out;                              // B*S*D f32
    float* weights = Sout + (size_t)B_N * S_LEN * D_DIM;         // B*S*S f32

    const size_t nX = (size_t)B_N * S_LEN * D_DIM;               // 8.39M
    const size_t nW = (size_t)D_DIM * D_DIM;                     // 262K
    uint16_t* xbf   = (uint16_t*)d_ws;
    uint16_t* Wqbf  = xbf + nX;
    uint16_t* Wkbf  = Wqbf + nW;
    uint16_t* Wvbf  = Wkbf + nW;
    uint16_t* Qbf   = Wvbf + nW;
    uint16_t* Kbf   = Qbf + nX;
    uint16_t* Vtbf  = Kbf + nX;                                  // [b][d][s]
    uint16_t* Wsbf  = Vtbf + nX;                                 // bf16 softmax weights

    // 0) one-shot f32->bf16 conversions
    cvt_bf16_kernel<<<dim3((int)(nX / 8 / 256)), dim3(256), 0, stream>>>(x, xbf, (int)(nX / 8));
    cvt_bf16_kernel<<<dim3((int)(nW / 8 / 256)), dim3(256), 0, stream>>>(Wq, Wqbf, (int)(nW / 8));
    cvt_bf16_kernel<<<dim3((int)(nW / 8 / 256)), dim3(256), 0, stream>>>(Wk, Wkbf, (int)(nW / 8));
    cvt_bf16_kernel<<<dim3((int)(nW / 8 / 256)), dim3(256), 0, stream>>>(Wv, Wvbf, (int)(nW / 8));

    const dim3 blk(128);

    // 1) projections (block tile 64x128)
    const dim3 g1(D_DIM / BN, (B_N * S_LEN) / BM);
    qkv_proj_kernel<<<g1, blk, 0, stream>>>(xbf, Wqbf, bq, Qbf, 0);
    qkv_proj_kernel<<<g1, blk, 0, stream>>>(xbf, Wkbf, bk, Kbf, 0);
    qkv_proj_kernel<<<g1, blk, 0, stream>>>(xbf, Wvbf, bv, Vtbf, 1);

    // 2) masked scaled scores -> f32 logits in d_out weights region
    const dim3 g2(S_LEN / BN, S_LEN / BM, B_N);
    scores_kernel<<<g2, blk, 0, stream>>>(Qbf, Kbf, mask, weights);

    // 3) in-place softmax + bf16 copy
    softmax_kernel<<<dim3(B_N * S_LEN), dim3(256), 0, stream>>>(weights, Wsbf);

    // 4) Sout = weights @ V
    const dim3 g4(D_DIM / BN, S_LEN / BM, B_N);
    out_kernel<<<g4, blk, 0, stream>>>(Wsbf, Vtbf, Sout);
}
